// GRULangModel_7464653160614
// MI455X (gfx1250) — matrix operand order repaired and checked by reading, not compiled, not run on hardware
//
#include <hip/hip_runtime.h>
#include <hip/hip_bf16.h>

// ---------------------------------------------------------------------------
// GRU language model on gfx1250 (CDNA5), wave32, v_wmma_f32_16x16x32_bf16,
// TDM (tensor_load_to_lds) double-buffered x_t staging, global_load_b128 via
// explicit address_space(1) pointers, hardware v_tanh_f32 / v_rcp_f32 gates.
// B=8, S=1024, VOCAB=32000, EMBED=256, HIDDEN=512.
// ---------------------------------------------------------------------------

#define B_    8
#define S_    1024
#define VOCAB 32000
#define EMB   256
#define HID   512
#define CAT   (HID + EMB)   // 768

typedef __attribute__((ext_vector_type(16))) __bf16 v16bf;
typedef __attribute__((ext_vector_type(8)))  float  v8f;
typedef __attribute__((ext_vector_type(4)))  unsigned int u32x4;
typedef __attribute__((ext_vector_type(4)))  unsigned int v4u;
typedef __attribute__((ext_vector_type(8)))  int v8i;
typedef __attribute__((ext_vector_type(4)))  int v4i;

// global (addrspace(1)) views so the backend emits global_load/store, not flat
typedef __attribute__((address_space(1))) __bf16       gbf;
typedef __attribute__((address_space(1))) const u32x4  gu4c;
typedef __attribute__((address_space(1))) float        gf32;

__device__ __forceinline__ const gbf* gcast(const __bf16* p) {
  return (const gbf*)(unsigned long long)p;
}
__device__ __forceinline__ gbf* gcast_mut(__bf16* p) {
  return (gbf*)(unsigned long long)p;
}

#if defined(__has_builtin)
#  if __has_builtin(__builtin_amdgcn_tensor_load_to_lds) && \
      __has_builtin(__builtin_amdgcn_s_wait_tensorcnt)
#    define HAVE_TDM 1
#  endif
#endif
#ifndef HAVE_TDM
#  define HAVE_TDM 0
#endif

// --- fast transcendental helpers (hardware TRANS ops on CDNA5) -------------
__device__ __forceinline__ float fast_rcp(float x) {
#if defined(__has_builtin) && __has_builtin(__builtin_amdgcn_rcpf)
  return __builtin_amdgcn_rcpf(x);
#else
  return 1.0f / x;
#endif
}

__device__ __forceinline__ float sigmoidf_(float x) {
  return fast_rcp(1.0f + __expf(-x));   // v_rcp_f32 + v_exp_f32
}

__device__ __forceinline__ float fast_tanh(float x) {
#if defined(__has_builtin) && __has_builtin(__builtin_amdgcn_tanhf)
  return __builtin_amdgcn_tanhf(x);     // v_tanh_f32
#else
  return 2.0f * sigmoidf_(2.0f * x) - 1.0f;
#endif
}

// --- WMMA wrapper: D = A(16x32 bf16) * B(32x16 bf16) + C(16x16 f32) ---------
__device__ __forceinline__ v8f bwmma(v16bf a, v16bf b, v8f c) {
  return __builtin_amdgcn_wmma_f32_16x16x32_bf16(
      false, a, false, b, (short)0, c, false, false);
}

// A fragment (16x32 bf16) from LDS/generic memory, leading dim ld.
__device__ __forceinline__ v16bf load_a_frag_l(const __bf16* base, int k0, int ld, int lane) {
  int r   = lane & 15;
  int sel = (lane >> 4) & 1;
  const __bf16* p = base + (size_t)r * ld + k0 + sel * 8;
  union { v16bf v; u32x4 u[2]; } x;
  x.u[0] = *(const u32x4*)(p);
  x.u[1] = *(const u32x4*)(p + 16);
  return x.v;
}

// A fragment from global memory.
__device__ __forceinline__ v16bf load_a_frag_g(const gbf* base, int row0, int k0, int ld, int lane) {
  int r   = lane & 15;
  int sel = (lane >> 4) & 1;
  const gbf* p = base + (size_t)(row0 + r) * ld + k0 + sel * 8;
  union { v16bf v; u32x4 u[2]; } x;
  x.u[0] = *(const gu4c*)(p);
  x.u[1] = *(const gu4c*)(p + 16);
  return x.v;
}

// B fragment (32x16 bf16) from global [K][N] row-major, leading dim ld.
__device__ __forceinline__ v16bf load_b_frag_g(const gbf* base, int k0, int n0, int ld, int lane) {
  int c   = lane & 15;
  int sel = (lane >> 4) & 1;
  const gbf* p = base + (size_t)(k0 + sel * 8) * ld + n0 + c;
  v16bf v;
  #pragma unroll
  for (int s = 0; s < 8; ++s) {
    v[s]     = p[(size_t)s * ld];
    v[8 + s] = p[(size_t)(16 + s) * ld];
  }
  return v;
}

#if HAVE_TDM
// TDM: async-copy nelem8*8 bytes (1-D tile, 8-byte elements) from global
// `src` into LDS byte offset `lds_off`. D# built per CDNA5 ISA ch.8.
__device__ __forceinline__ void tdm_load_1d(const __bf16* src, unsigned lds_off,
                                            unsigned nelem8) {
  unsigned long long ga = (unsigned long long)src;
  v4u g0;
  g0.x = 1u;                                        // count=1, user mode
  g0.y = lds_off;                                   // LDS dest (bytes)
  g0.z = (unsigned)ga;                              // global_addr[31:0]
  g0.w = ((unsigned)(ga >> 32) & 0x1FFFFFFu) | (2u << 30);  // addr[56:32], type=2
  v8i g1;
  g1[0] = 3 << 16;                                  // data_size=3 (8B), mask=0
  g1[1] = (int)((nelem8 & 0xFFFFu) << 16);          // tensor_dim0[15:0]
  g1[2] = (int)((nelem8 >> 16) | (1u << 16));       // tensor_dim0[31:16], tensor_dim1=1
  g1[3] = (int)((nelem8 & 0xFFFFu) << 16);          // tile_dim0
  g1[4] = 1;                                        // tile_dim1=1, tile_dim2=0
  g1[5] = (int)nelem8;                              // tensor_dim0_stride[31:0]
  g1[6] = 0;
  g1[7] = 0;
  v4i gz = {0, 0, 0, 0};
#if __clang_major__ >= 23
  v8i gz8 = {0, 0, 0, 0, 0, 0, 0, 0};
  __builtin_amdgcn_tensor_load_to_lds(g0, g1, gz, gz, gz8, 0);
#else
  __builtin_amdgcn_tensor_load_to_lds(g0, g1, gz, gz, 0);
#endif
}
#endif

// --- f32 -> bf16 conversion ------------------------------------------------
__global__ void f32_to_bf16_kernel(const float* __restrict__ in,
                                   __bf16* __restrict__ out, int n) {
  int i = blockIdx.x * blockDim.x + threadIdx.x;
  if (i < n) out[i] = (__bf16)in[i];
}

// --- embedding gather -> x_bf laid out [S][B][EMB] -------------------------
__global__ void embed_gather_kernel(const int* __restrict__ tok,
                                    const float* __restrict__ embed,
                                    __bf16* __restrict__ xbf) {
  int i = blockIdx.x * blockDim.x + threadIdx.x;   // over B*S*EMB
  if (i >= B_ * S_ * EMB) return;
  int e  = i % EMB;
  int bs = i / EMB;
  int b  = bs / S_;
  int s  = bs % S_;
  int t  = tok[b * S_ + s];
  xbf[((size_t)s * B_ + b) * EMB + e] = (__bf16)embed[(size_t)t * EMB + e];
}

// --- persistent GRU recurrence (1 workgroup, 16 waves) ---------------------
__global__ __launch_bounds__(512) void gru_kernel(
    const __bf16* __restrict__ xbf,
    const __bf16* __restrict__ Wr, const __bf16* __restrict__ Wz,
    const __bf16* __restrict__ Wc,
    const float* __restrict__ br, const float* __restrict__ bz,
    const float* __restrict__ bc,
    __bf16* __restrict__ hs) {
  __shared__ __bf16 h_bf[16 * HID];       // padded 16 rows; rows 8..15 stay 0
  __shared__ __bf16 rh_bf[16 * HID];      // r*h, padded
  __shared__ __bf16 x_bf[2][16 * EMB];    // double-buffered x_t, padded

  const int tid  = threadIdx.x;
  const int lane = tid & 31;
  const int wave = tid >> 5;           // 0..15
  const int n0   = wave * 32;          // this wave's 32 hidden columns
  const int nc   = lane & 15;
  const int hi   = lane >> 4;

  for (int i = tid; i < 16 * HID; i += 512) { h_bf[i] = (__bf16)0.0f; rh_bf[i] = (__bf16)0.0f; }
  {
    __bf16* xz = &x_bf[0][0];
    for (int i = tid; i < 2 * 16 * EMB; i += 512) xz[i] = (__bf16)0.0f;
  }
  __syncthreads();

  // f32 hidden state lives in registers: lane (hi==0) owns rows i=0..7 at
  // columns n0+nc (tile0) and n0+16+nc (tile1).
  float hr0[8] = {}, hr1[8] = {};

  const float brv0 = br[n0 + nc],      bzv0 = bz[n0 + nc],      bcv0 = bc[n0 + nc];
  const float brv1 = br[n0 + 16 + nc], bzv1 = bz[n0 + 16 + nc], bcv1 = bc[n0 + 16 + nc];

  gbf* hsg = gcast_mut(hs);

#if HAVE_TDM
  // prologue: DMA x_0 into buffer 0 (4 KB = 512 x 8B)
  if (wave == 0) {
    tdm_load_1d(xbf, (unsigned)(unsigned long long)&x_bf[0][0], 512u);
    __builtin_amdgcn_s_wait_tensorcnt((short)0);
  }
  __syncthreads();
#endif

  for (int t = 0; t < S_; ++t) {
    // Launder a zero scalar offset so weight-fragment loads cannot be hoisted
    // (and spilled) out of the time loop; weights stream from L2 each step.
    int zo = 0;
    asm volatile("" : "+s"(zo));
    const gbf* WrT = gcast(Wr) + zo;
    const gbf* WzT = gcast(Wz) + zo;
    const gbf* WcT = gcast(Wc) + zo;

#if HAVE_TDM
    const __bf16* xcur = x_bf[t & 1];
    if (wave == 0 && t + 1 < S_) {     // async-stage next x_t via TDM
      tdm_load_1d(xbf + (size_t)(t + 1) * B_ * EMB,
                  (unsigned)(unsigned long long)&x_bf[(t + 1) & 1][0], 512u);
    }
#else
    const __bf16* xcur = x_bf[0];
    const __bf16* xt = xbf + (size_t)t * B_ * EMB;
    for (int i = tid; i < B_ * EMB; i += 512) x_bf[0][i] = xt[i];
    if (wave == 0 && t + 1 < S_) __builtin_prefetch(xt + B_ * EMB + lane * 64, 0, 0);
    __syncthreads();
#endif

    // ---- gates r and z on this wave's 32 columns ----
    v8f ar0 = {}, ar1 = {}, az0 = {}, az1 = {};
    #pragma unroll 1
    for (int k0 = 0; k0 < HID; k0 += 32) {          // h part of catted
      v16bf a  = load_a_frag_l(h_bf, k0, HID, lane);
      v16bf b0 = load_b_frag_g(WrT, k0, n0,      HID, lane);
      v16bf b1 = load_b_frag_g(WrT, k0, n0 + 16, HID, lane);
      v16bf b2 = load_b_frag_g(WzT, k0, n0,      HID, lane);
      v16bf b3 = load_b_frag_g(WzT, k0, n0 + 16, HID, lane);
      ar0 = bwmma(a, b0, ar0);
      ar1 = bwmma(a, b1, ar1);
      az0 = bwmma(a, b2, az0);
      az1 = bwmma(a, b3, az1);
    }
    #pragma unroll 1
    for (int k0 = 0; k0 < EMB; k0 += 32) {          // x part of catted
      v16bf a  = load_a_frag_l(xcur, k0, EMB, lane);
      v16bf b0 = load_b_frag_g(WrT, HID + k0, n0,      HID, lane);
      v16bf b1 = load_b_frag_g(WrT, HID + k0, n0 + 16, HID, lane);
      v16bf b2 = load_b_frag_g(WzT, HID + k0, n0,      HID, lane);
      v16bf b3 = load_b_frag_g(WzT, HID + k0, n0 + 16, HID, lane);
      ar0 = bwmma(a, b0, ar0);
      ar1 = bwmma(a, b1, ar1);
      az0 = bwmma(a, b2, az0);
      az1 = bwmma(a, b3, az1);
    }

    // ---- r = sigmoid(.); stage r*h (bf16). Rows 8..15 remain zero. ----
    if (hi == 0) {
      #pragma unroll
      for (int i = 0; i < 8; ++i) {
        float rv0 = sigmoidf_(ar0[i] + brv0);
        float rv1 = sigmoidf_(ar1[i] + brv1);
        rh_bf[i * HID + n0 + nc]      = (__bf16)(rv0 * hr0[i]);
        rh_bf[i * HID + n0 + 16 + nc] = (__bf16)(rv1 * hr1[i]);
      }
    }
    __syncthreads();

    // ---- candidate: [r*h, x] @ Wc ----
    v8f ac0 = {}, ac1 = {};
    #pragma unroll 1
    for (int k0 = 0; k0 < HID; k0 += 32) {
      v16bf a  = load_a_frag_l(rh_bf, k0, HID, lane);
      v16bf b0 = load_b_frag_g(WcT, k0, n0,      HID, lane);
      v16bf b1 = load_b_frag_g(WcT, k0, n0 + 16, HID, lane);
      ac0 = bwmma(a, b0, ac0);
      ac1 = bwmma(a, b1, ac1);
    }
    #pragma unroll 1
    for (int k0 = 0; k0 < EMB; k0 += 32) {
      v16bf a  = load_a_frag_l(xcur, k0, EMB, lane);
      v16bf b0 = load_b_frag_g(WcT, HID + k0, n0,      HID, lane);
      v16bf b1 = load_b_frag_g(WcT, HID + k0, n0 + 16, HID, lane);
      ac0 = bwmma(a, b0, ac0);
      ac1 = bwmma(a, b1, ac1);
    }

    // ---- h_new = (1-z)*tanh(c) + z*h ; update state + stream to hs ----
    if (hi == 0) {
      #pragma unroll
      for (int i = 0; i < 8; ++i) {
        float z0 = sigmoidf_(az0[i] + bzv0);
        float z1 = sigmoidf_(az1[i] + bzv1);
        float c0 = fast_tanh(ac0[i] + bcv0);
        float c1 = fast_tanh(ac1[i] + bcv1);
        float hn0 = (1.0f - z0) * c0 + z0 * hr0[i];
        float hn1 = (1.0f - z1) * c1 + z1 * hr1[i];
        hr0[i] = hn0;
        hr1[i] = hn1;
        h_bf[i * HID + n0 + nc]      = (__bf16)hn0;
        h_bf[i * HID + n0 + 16 + nc] = (__bf16)hn1;
        hsg[((size_t)i * S_ + t) * HID + n0 + nc]      = (__bf16)hn0;
        hsg[((size_t)i * S_ + t) * HID + n0 + 16 + nc] = (__bf16)hn1;
      }
    }
#if HAVE_TDM
    if (wave == 0) __builtin_amdgcn_s_wait_tensorcnt((short)0);
#endif
    __syncthreads();   // h_bf/rh/x consumers done before next step overwrites
  }
}

// --- output projection: [8192x512] @ [512x32000] + bo, f32 out -------------
// Block tile 128x128, 8 waves arranged 2(M) x 4(N); wave tile 64x32.
__global__ __launch_bounds__(256) void out_gemm_kernel(
    const __bf16* __restrict__ hs_, const __bf16* __restrict__ Wo_,
    const float* __restrict__ bo, float* __restrict__ out_) {
  const gbf* hs = gcast(hs_);
  const gbf* Wo = gcast(Wo_);
  gf32* out = (gf32*)(unsigned long long)out_;

  const int tid  = threadIdx.x;
  const int lane = tid & 31;
  const int wave = tid >> 5;
  const int mw   = wave & 1;        // 2 wave-rows (M)
  const int nw   = wave >> 1;       // 4 wave-cols (N)
  const int m0   = blockIdx.y * 128 + mw * 64;   // wave covers 64 rows
  const int n0   = blockIdx.x * 128 + nw * 32;   // wave covers 32 cols

  v8f acc[4][2];
  #pragma unroll
  for (int i = 0; i < 4; ++i)
    #pragma unroll
    for (int j = 0; j < 2; ++j) acc[i][j] = (v8f){};

  #pragma unroll 1
  for (int k0 = 0; k0 < HID; k0 += 32) {
    v16bf a0 = load_a_frag_g(hs, m0,      k0, HID, lane);
    v16bf a1 = load_a_frag_g(hs, m0 + 16, k0, HID, lane);
    v16bf a2 = load_a_frag_g(hs, m0 + 32, k0, HID, lane);
    v16bf a3 = load_a_frag_g(hs, m0 + 48, k0, HID, lane);
    v16bf b0 = load_b_frag_g(Wo, k0, n0,      VOCAB, lane);
    v16bf b1 = load_b_frag_g(Wo, k0, n0 + 16, VOCAB, lane);
    acc[0][0] = bwmma(a0, b0, acc[0][0]);
    acc[0][1] = bwmma(a0, b1, acc[0][1]);
    acc[1][0] = bwmma(a1, b0, acc[1][0]);
    acc[1][1] = bwmma(a1, b1, acc[1][1]);
    acc[2][0] = bwmma(a2, b0, acc[2][0]);
    acc[2][1] = bwmma(a2, b1, acc[2][1]);
    acc[3][0] = bwmma(a3, b0, acc[3][0]);
    acc[3][1] = bwmma(a3, b1, acc[3][1]);
  }

  const int nc = lane & 15;
  const int hi = lane >> 4;
  const float bias0 = bo[n0 + nc];
  const float bias1 = bo[n0 + 16 + nc];
  #pragma unroll
  for (int i = 0; i < 4; ++i) {
    #pragma unroll
    for (int r = 0; r < 8; ++r) {
      int m = m0 + i * 16 + r + 8 * hi;
      out[(size_t)m * VOCAB + n0 + nc]      = acc[i][0][r] + bias0;
      out[(size_t)m * VOCAB + n0 + 16 + nc] = acc[i][1][r] + bias1;
    }
  }
}

// ---------------------------------------------------------------------------
extern "C" void kernel_launch(void* const* d_in, const int* in_sizes, int n_in,
                              void* d_out, int out_size, void* d_ws, size_t ws_size,
                              hipStream_t stream) {
  const int*   tok   = (const int*)d_in[0];
  const float* embed = (const float*)d_in[1];
  const float* Wr_f  = (const float*)d_in[2];
  const float* br    = (const float*)d_in[3];
  const float* Wz_f  = (const float*)d_in[4];
  const float* bz    = (const float*)d_in[5];
  const float* Wc_f  = (const float*)d_in[6];
  const float* bc    = (const float*)d_in[7];
  const float* Wo_f  = (const float*)d_in[8];
  const float* bo    = (const float*)d_in[9];
  float* out = (float*)d_out;

  // workspace layout (256B aligned)
  char*  ws  = (char*)d_ws;
  size_t off = 0;
  auto carve = [&](size_t bytes) -> char* {
    char* p = ws + off;
    off = (off + bytes + 255) & ~(size_t)255;
    return p;
  };
  __bf16* xbf   = (__bf16*)carve((size_t)B_ * S_ * EMB * 2);   //  4.0 MB
  __bf16* Wr_bf = (__bf16*)carve((size_t)CAT * HID * 2);       //  0.75 MB
  __bf16* Wz_bf = (__bf16*)carve((size_t)CAT * HID * 2);
  __bf16* Wc_bf = (__bf16*)carve((size_t)CAT * HID * 2);
  __bf16* Wo_bf = (__bf16*)carve((size_t)HID * VOCAB * 2);     // 32.8 MB
  __bf16* hs_bf = (__bf16*)carve((size_t)B_ * S_ * HID * 2);   //  8.4 MB

  // 1) weight conversions to bf16
  {
    int n = CAT * HID;
    f32_to_bf16_kernel<<<(n + 255) / 256, 256, 0, stream>>>(Wr_f, Wr_bf, n);
    f32_to_bf16_kernel<<<(n + 255) / 256, 256, 0, stream>>>(Wz_f, Wz_bf, n);
    f32_to_bf16_kernel<<<(n + 255) / 256, 256, 0, stream>>>(Wc_f, Wc_bf, n);
  }
  {
    int n = HID * VOCAB;
    f32_to_bf16_kernel<<<(n + 255) / 256, 256, 0, stream>>>(Wo_f, Wo_bf, n);
  }

  // 2) embedding gather into [S][B][EMB] bf16
  {
    int n = B_ * S_ * EMB;
    embed_gather_kernel<<<(n + 255) / 256, 256, 0, stream>>>(tok, embed, xbf);
  }

  // 3) sequential GRU recurrence (persistent single workgroup, 16 waves)
  gru_kernel<<<1, 512, 0, stream>>>(xbf, Wr_bf, Wz_bf, Wc_bf, br, bz, bc, hs_bf);

  // 4) output projection GEMM -> f32 logits
  dim3 grid(VOCAB / 128, (B_ * S_) / 128);   // 250 x 64
  out_gemm_kernel<<<grid, 256, 0, stream>>>(hs_bf, Wo_bf, bo, out);
}
